// Encoder_62723702391354
// MI455X (gfx1250) — matrix operand order repaired and checked by reading, not compiled, not run on hardware
//
#include <hip/hip_runtime.h>

// ---------------------------------------------------------------------------
// GATv2 x2 layers for MI455X (gfx1250, wave32, WMMA).
// GEMMs: bf16 WMMA 16x16x32 with async global->LDS double buffering.
// Edge softmax: atomics-based segmented softmax.
// ---------------------------------------------------------------------------

typedef __attribute__((ext_vector_type(16))) __bf16 v16bf;
typedef __attribute__((ext_vector_type(8)))  float  v8f;
typedef __attribute__((__vector_size__(4 * sizeof(int)))) int v4i;

union FragBF { v16bf v; uint4 q[2]; unsigned short u[16]; };
union AccF   { v8f v; float f[8]; };

#if __has_builtin(__builtin_amdgcn_global_load_async_to_lds_b128) && \
    __has_builtin(__builtin_amdgcn_s_wait_asynccnt)
#define USE_ASYNC_LDS 1
#else
#define USE_ASYNC_LDS 0
#endif

#define AS1 __attribute__((address_space(1)))
#define AS3 __attribute__((address_space(3)))

__device__ __forceinline__ AS1 v4i* as_global_v4(const void* p) {
    return (AS1 v4i*)(unsigned long long)(uintptr_t)p;
}
// Generic->LDS: flat LDS aperture address bits [31:0] are the LDS offset.
__device__ __forceinline__ AS3 v4i* as_lds_v4(const void* p) {
    return (AS3 v4i*)(unsigned)(uintptr_t)p;
}

__device__ __forceinline__ unsigned short f2bf(float f) {
    union { float f; unsigned u; } c; c.f = f;
    unsigned u = c.u;
    u += 0x7FFFu + ((u >> 16) & 1u);        // round-to-nearest-even
    return (unsigned short)(u >> 16);
}
__device__ __forceinline__ unsigned fkey(float f) {   // monotone float->uint
    union { float f; unsigned u; } c; c.f = f;
    return c.u ^ ((c.u >> 31) ? 0xFFFFFFFFu : 0x80000000u);
}
__device__ __forceinline__ float fdec(unsigned k) {
    union { unsigned u; float f; } c;
    c.u = k ^ ((k >> 31) ? 0x80000000u : 0xFFFFFFFFu);
    return c.f;
}

// --------------------------- conversion kernels ----------------------------

__global__ void convert_bf16(const float* __restrict__ in,
                             unsigned short* __restrict__ out, long n) {
    long i = (long)blockIdx.x * blockDim.x + threadIdx.x;
    if (i < n) out[i] = f2bf(in[i]);
}

// Wt[n][k] = bf16(W[k][n])   (W is [K, Nc] row-major)
__global__ void convert_wt(const float* __restrict__ W,
                           unsigned short* __restrict__ Wt, int K, int Nc) {
    long i = (long)blockIdx.x * blockDim.x + threadIdx.x;
    if (i >= (long)K * Nc) return;
    int k = (int)(i % K);
    int n = (int)(i / K);
    Wt[i] = f2bf(W[(size_t)k * Nc + n]);
}

__global__ void fill_zero(float* __restrict__ p, long n) {
    long i = (long)blockIdx.x * blockDim.x + threadIdx.x;
    if (i < n) p[i] = 0.0f;
}

// ------------------------------ WMMA GEMM ----------------------------------
// C[M, Nc] = bf16(A[M, K]) @ bf16(W[K, Nc]) + bias ; W supplied transposed.
// Block: 256 threads (8 waves). Tile: 64 rows x 128 cols. K-step 32.
// Double-buffered LDS, filled by GLOBAL_LOAD_ASYNC_TO_LDS_B128 when available.

#define BM 64
#define BN 128
#define BK 32
#define LDA 48   // BK + 16 pad (96 B rows: 16-B aligned)
#define LDB 48

__global__ __launch_bounds__(256) void gemm_bf16_wmma(
    const unsigned short* __restrict__ A,   // [M, K] bf16 bits
    const unsigned short* __restrict__ Wt,  // [Nc, K] bf16 bits (W transposed)
    const float* __restrict__ bias,         // [Nc]
    float* __restrict__ C,                  // [M, Nc]
    int M, int K, int Nc)
{
    __shared__ unsigned short lA[2][BM][LDA];
    __shared__ unsigned short lB[2][BN][LDB];

    const int tid  = threadIdx.x;
    const int lane = tid & 31;
    const int wid  = tid >> 5;       // 0..7
    const int wm   = wid & 3;        // 16-row strip within tile
    const int wn   = wid >> 2;       // 0..1 : 64-col strip
    const int rowBlk = blockIdx.x * BM;
    const int colBlk = blockIdx.y * BN;

    const int half = lane >> 4;
    const int l15  = lane & 15;

    AccF acc[4];
    #pragma unroll
    for (int t = 0; t < 4; ++t) {
        float b = bias[colBlk + wn * 64 + t * 16 + l15];
        #pragma unroll
        for (int r = 0; r < 8; ++r) acc[t].f[r] = b;
    }

    // global->LDS tile-load indexing (per thread)
    const int arow = tid >> 2;            // 0..63
    const int acol = (tid & 3) * 8;       // 0,8,16,24  (8 bf16 = 16 B)
    const int brow = tid >> 1;            // 0..127
    const int bcol = (tid & 1) * 16;      // 0,16       (2 x 16 B)

    // Clamp OOB A rows to a valid row: rows >= M are computed but never stored.
    int grA = rowBlk + arow; if (grA >= M) grA = M - 1;
    const unsigned short* gA = A  + (size_t)grA * K + acol;
    const unsigned short* gB = Wt + (size_t)(colBlk + brow) * K + bcol;

#if USE_ASYNC_LDS
    // ---- async double-buffered pipeline ----
    {   // prologue: fill stage 0
        __builtin_amdgcn_global_load_async_to_lds_b128(
            as_global_v4(gA), as_lds_v4(&lA[0][arow][acol]), 0, 0);
        __builtin_amdgcn_global_load_async_to_lds_b128(
            as_global_v4(gB), as_lds_v4(&lB[0][brow][bcol]), 0, 0);
        __builtin_amdgcn_global_load_async_to_lds_b128(
            as_global_v4(gB + 8), as_lds_v4(&lB[0][brow][bcol + 8]), 0, 0);
    }
    for (int k0 = 0; k0 < K; k0 += BK) {
        const int p = (k0 / BK) & 1;
        __builtin_amdgcn_s_wait_asynccnt(0);   // my stage-p fills done
        __syncthreads();                       // everyone's stage-p fills done
        if (k0 + BK < K) {                     // prefetch stage 1-p
            const int q = 1 - p;
            __builtin_amdgcn_global_load_async_to_lds_b128(
                as_global_v4(gA + k0 + BK), as_lds_v4(&lA[q][arow][acol]), 0, 0);
            __builtin_amdgcn_global_load_async_to_lds_b128(
                as_global_v4(gB + k0 + BK), as_lds_v4(&lB[q][brow][bcol]), 0, 0);
            __builtin_amdgcn_global_load_async_to_lds_b128(
                as_global_v4(gB + k0 + BK + 8), as_lds_v4(&lB[q][brow][bcol + 8]), 0, 0);
        }
        // A fragment (ISA 16-bit A 16x32 layout):
        //   lane l: Mrow = l&15 ; frag[0..7]  = A[Mrow][half*8 .. +8)
        //                         frag[8..15] = A[Mrow][16+half*8 .. +8)
        FragBF a;
        {
            const unsigned short* pa = &lA[p][wm * 16 + l15][half * 8];
            a.q[0] = *(const uint4*)pa;
            a.q[1] = *(const uint4*)(pa + 16);
        }
        #pragma unroll
        for (int t = 0; t < 4; ++t) {
            // B fragment (32x16): lane l holds col = l&15,
            //   frag[0..7] = B[half*8 .. +8)[col], frag[8..15] = B[16+half*8 .. +8)[col]  (LDS holds W^T rows)
            FragBF b;
            const unsigned short* pb = &lB[p][wn * 64 + t * 16 + l15][half * 8];
            b.q[0] = *(const uint4*)pb;
            b.q[1] = *(const uint4*)(pb + 16);
            acc[t].v = __builtin_amdgcn_wmma_f32_16x16x32_bf16(
                false, a.v, false, b.v, (short)0, acc[t].v, false, false);
        }
    }
#else
    // ---- fallback: synchronous single-stage staging ----
    for (int k0 = 0; k0 < K; k0 += BK) {
        {
            uint4 v = *(const uint4*)(gA + k0);
            *(uint4*)&lA[0][arow][acol] = v;
        }
        {
            *(uint4*)&lB[0][brow][bcol]     = *(const uint4*)(gB + k0);
            *(uint4*)&lB[0][brow][bcol + 8] = *(const uint4*)(gB + k0 + 8);
        }
        __syncthreads();
        FragBF a;
        {
            const unsigned short* pa = &lA[0][wm * 16 + l15][half * 8];
            a.q[0] = *(const uint4*)pa;
            a.q[1] = *(const uint4*)(pa + 16);
        }
        #pragma unroll
        for (int t = 0; t < 4; ++t) {
            FragBF b;
            const unsigned short* pb = &lB[0][wn * 64 + t * 16 + l15][half * 8];
            b.q[0] = *(const uint4*)pb;
            b.q[1] = *(const uint4*)(pb + 16);
            acc[t].v = __builtin_amdgcn_wmma_f32_16x16x32_bf16(
                false, a.v, false, b.v, (short)0, acc[t].v, false, false);
        }
        __syncthreads();
    }
#endif

    // C/D layout: VGPR r -> row = wm*16 + half*8 + r, col = l&15
    #pragma unroll
    for (int t = 0; t < 4; ++t) {
        int col = colBlk + wn * 64 + t * 16 + l15;
        #pragma unroll
        for (int r = 0; r < 8; ++r) {
            int row = rowBlk + wm * 16 + half * 8 + r;
            if (row < M) C[(size_t)row * Nc + col] = acc[t].f[r];
        }
    }
}

// ------------------------------ edge kernels -------------------------------
// One wave32 per edge. H = 4 heads fixed; C = per-head channels (64 or 128).

template<int C>
__global__ __launch_bounds__(256) void edge_scores(
    const int* __restrict__ ei, int E, int Etot,
    const float* __restrict__ xl, const float* __restrict__ xr,
    const float* __restrict__ att,          // [4*C]
    float* __restrict__ ebuf,               // [Etot*4]
    unsigned* __restrict__ mbuf)            // [N*4], monotone-uint max
{
    int wid = threadIdx.x >> 5, lane = threadIdx.x & 31;
    int eid = blockIdx.x * 8 + wid;
    if (eid >= Etot) return;
    int src, dst;
    if (eid < E) { src = ei[eid]; dst = ei[E + eid]; }
    else         { src = dst = eid - E; }

    const float* pl = xl + (size_t)src * (4 * C);
    const float* pr = xr + (size_t)dst * (4 * C);
    #pragma unroll
    for (int h = 0; h < 4; ++h) {
        float p = 0.0f;
        for (int c = lane; c < C; c += 32) {
            float v = pl[h * C + c] + pr[h * C + c];
            v = v > 0.0f ? v : 0.2f * v;              // leaky_relu
            p += v * att[h * C + c];
        }
        #pragma unroll
        for (int off = 16; off; off >>= 1) p += __shfl_down(p, off, 32);
        if (lane == 0) {
            ebuf[(size_t)eid * 4 + h] = p;
            atomicMax(&mbuf[dst * 4 + h], fkey(p));
        }
    }
}

__global__ void edge_expsum(const int* __restrict__ ei, int E, int Etot,
                            float* __restrict__ ebuf,
                            const unsigned* __restrict__ mbuf,
                            float* __restrict__ denom) {
    long t = (long)blockIdx.x * blockDim.x + threadIdx.x;
    if (t >= (long)Etot * 4) return;
    int eid = (int)(t >> 2), h = (int)(t & 3);
    int dst = (eid < E) ? ei[E + eid] : (eid - E);
    float ex = __expf(ebuf[t] - fdec(mbuf[dst * 4 + h]));
    ebuf[t] = ex;
    atomicAdd(&denom[dst * 4 + h], ex);
}

template<int C>
__global__ __launch_bounds__(256) void edge_aggr(
    const int* __restrict__ ei, int E, int Etot,
    const float* __restrict__ ebuf, const float* __restrict__ denom,
    const float* __restrict__ xl, float* __restrict__ outAcc) {
    int wid = threadIdx.x >> 5, lane = threadIdx.x & 31;
    int eid = blockIdx.x * 8 + wid;
    if (eid >= Etot) return;
    int src, dst;
    if (eid < E) { src = ei[eid]; dst = ei[E + eid]; }
    else         { src = dst = eid - E; }
    const int HC = 4 * C;
    for (int hc = lane; hc < HC; hc += 32) {
        int h = hc / C;
        float alpha = ebuf[(size_t)eid * 4 + h] / (denom[dst * 4 + h] + 1e-16f);
        atomicAdd(&outAcc[(size_t)dst * HC + hc],
                  xl[(size_t)src * HC + hc] * alpha);
    }
}

// ----------------------------- finalize kernels ----------------------------

__global__ void finalize1(const float* __restrict__ out1,
                          const float* __restrict__ bias,
                          unsigned short* __restrict__ hbf, long n) {
    long i = (long)blockIdx.x * blockDim.x + threadIdx.x;
    if (i >= n) return;
    float v = out1[i] + bias[i & 255];
    v = v > 0.0f ? v : 0.0f;                          // relu
    hbf[i] = f2bf(v);
}

__global__ void finalize2(const float* __restrict__ out2,
                          const float* __restrict__ bias2,
                          float* __restrict__ y, long n) {
    long i = (long)blockIdx.x * blockDim.x + threadIdx.x;
    if (i >= n) return;
    int  c  = (int)(i & 127);
    long nd = i >> 7;
    const float* p = out2 + nd * 512;
    float s = p[c] + p[128 + c] + p[256 + c] + p[384 + c];
    y[i] = 0.25f * s + bias2[c];                      // mean over heads + bias
}

// -------------------------------- launcher ---------------------------------

extern "C" void kernel_launch(void* const* d_in, const int* in_sizes, int n_in,
                              void* d_out, int out_size, void* d_ws, size_t ws_size,
                              hipStream_t stream) {
    (void)n_in; (void)out_size; (void)ws_size;

    const float* x     = (const float*)d_in[0];
    const int*   ei    = (const int*)d_in[1];       // harness contract: int32
    const float* W1l   = (const float*)d_in[2];
    const float* b1l   = (const float*)d_in[3];
    const float* W1r   = (const float*)d_in[4];
    const float* b1r   = (const float*)d_in[5];
    const float* att1  = (const float*)d_in[6];
    const float* bias1 = (const float*)d_in[7];
    const float* W2l   = (const float*)d_in[8];
    const float* b2l   = (const float*)d_in[9];
    const float* W2r   = (const float*)d_in[10];
    const float* b2r   = (const float*)d_in[11];
    const float* att2  = (const float*)d_in[12];
    const float* bias2 = (const float*)d_in[13];
    float* y = (float*)d_out;

    const int IND = 512, K1 = 256, K2 = 512;        // K1=H*HID, K2=H*2*OUT
    const int Nn   = in_sizes[0] / IND;
    const int E    = in_sizes[1] / 2;
    const int Etot = E + Nn;

    // ---- workspace layout (phase-reused) ----
    char* ws = (char*)d_ws;
    size_t off = 0;
    auto alloc = [&](size_t bytes) -> char* {
        char* p = ws + off;
        off += (bytes + 255) & ~(size_t)255;
        return p;
    };
    unsigned short* WT1l = (unsigned short*)alloc((size_t)IND * K1 * 2);
    unsigned short* WT1r = (unsigned short*)alloc((size_t)IND * K1 * 2);
    unsigned short* WT2l = (unsigned short*)alloc((size_t)K1 * K2 * 2);
    unsigned short* WT2r = (unsigned short*)alloc((size_t)K1 * K2 * 2);
    unsigned short* xbf  = (unsigned short*)alloc((size_t)Nn * IND * 2); // -> h bf16
    float* P1   = (float*)alloc((size_t)Nn * K2 * 4);  // xl1|xr1 -> xl2
    float* P2   = (float*)alloc((size_t)Nn * K2 * 4);  // out1    -> xr2
    float* P3   = (float*)alloc((size_t)Nn * K2 * 4);  // out2 accum
    float* ebuf = (float*)alloc((size_t)Etot * 4 * 4);
    unsigned* mbuf  = (unsigned*)alloc((size_t)Nn * 4 * 4);
    float*    denom = (float*)alloc((size_t)Nn * 4 * 4);

    float* xl1  = P1;
    float* xr1  = P1 + (size_t)Nn * K1;
    float* out1 = P2;
    unsigned short* hbf = xbf;
    float* xl2  = P1;
    float* xr2  = P2;
    float* out2 = P3;

    const int TB = 256;
    auto blks = [](long n, int tb) { return (int)((n + tb - 1) / tb); };

    // ---- weight transpose + bf16 convert ----
    convert_wt<<<blks((long)IND * K1, TB), TB, 0, stream>>>(W1l, WT1l, IND, K1);
    convert_wt<<<blks((long)IND * K1, TB), TB, 0, stream>>>(W1r, WT1r, IND, K1);
    convert_wt<<<blks((long)K1 * K2, TB), TB, 0, stream>>>(W2l, WT2l, K1, K2);
    convert_wt<<<blks((long)K1 * K2, TB), TB, 0, stream>>>(W2r, WT2r, K1, K2);
    convert_bf16<<<blks((long)Nn * IND, TB), TB, 0, stream>>>(x, xbf, (long)Nn * IND);

    // ---- layer 1 GEMMs (512 -> 256) ----
    dim3 g1((Nn + BM - 1) / BM, K1 / BN);
    gemm_bf16_wmma<<<g1, TB, 0, stream>>>(xbf, WT1l, b1l, xl1, Nn, IND, K1);
    gemm_bf16_wmma<<<g1, TB, 0, stream>>>(xbf, WT1r, b1r, xr1, Nn, IND, K1);

    // ---- layer 1 segmented softmax + aggregate ----
    fill_zero<<<blks((long)Nn * 4, TB), TB, 0, stream>>>((float*)mbuf, (long)Nn * 4);
    fill_zero<<<blks((long)Nn * 4, TB), TB, 0, stream>>>(denom, (long)Nn * 4);
    fill_zero<<<blks((long)Nn * K1, TB), TB, 0, stream>>>(out1, (long)Nn * K1);
    int egrid = (Etot + 7) / 8;
    edge_scores<64><<<egrid, TB, 0, stream>>>(ei, E, Etot, xl1, xr1, att1, ebuf, mbuf);
    edge_expsum<<<blks((long)Etot * 4, TB), TB, 0, stream>>>(ei, E, Etot, ebuf, mbuf, denom);
    edge_aggr<64><<<egrid, TB, 0, stream>>>(ei, E, Etot, ebuf, denom, xl1, out1);
    finalize1<<<blks((long)Nn * K1, TB), TB, 0, stream>>>(out1, bias1, hbf, (long)Nn * K1);

    // ---- layer 2 GEMMs (256 -> 512) ----
    dim3 g2((Nn + BM - 1) / BM, K2 / BN);
    gemm_bf16_wmma<<<g2, TB, 0, stream>>>(hbf, WT2l, b2l, xl2, Nn, K1, K2);
    gemm_bf16_wmma<<<g2, TB, 0, stream>>>(hbf, WT2r, b2r, xr2, Nn, K1, K2);

    // ---- layer 2 segmented softmax + aggregate ----
    fill_zero<<<blks((long)Nn * 4, TB), TB, 0, stream>>>((float*)mbuf, (long)Nn * 4);
    fill_zero<<<blks((long)Nn * 4, TB), TB, 0, stream>>>(denom, (long)Nn * 4);
    fill_zero<<<blks((long)Nn * K2, TB), TB, 0, stream>>>(out2, (long)Nn * K2);
    edge_scores<128><<<egrid, TB, 0, stream>>>(ei, E, Etot, xl2, xr2, att2, ebuf, mbuf);
    edge_expsum<<<blks((long)Etot * 4, TB), TB, 0, stream>>>(ei, E, Etot, ebuf, mbuf, denom);
    edge_aggr<128><<<egrid, TB, 0, stream>>>(ei, E, Etot, ebuf, denom, xl2, out2);

    // ---- mean over heads + bias -> d_out ----
    finalize2<<<blks((long)Nn * 128, TB), TB, 0, stream>>>(out2, bias2, y, (long)Nn * 128);
}
